// MoETransformerModel_52450140618934
// MI455X (gfx1250) — matrix operand order repaired and checked by reading, not compiled, not run on hardware
//
#include <hip/hip_runtime.h>

// ---------------- model constants ----------------
constexpr int V_  = 50257;
constexpr int D_  = 1024;
constexpr int NH_ = 16;
constexpr int FF_ = 4096;
constexpr int E_  = 16;
constexpr int O_  = 1000;
constexpr int B_  = 256;
constexpr int S_  = 128;
constexpr int DH_ = D_ / NH_;          // 64
constexpr int MTOK = B_ * S_;          // 32768

// ---------------- WMMA types (CDNA5 gfx1250, wave32) ----------------
typedef __bf16 bf16_t;
typedef __attribute__((ext_vector_type(16))) bf16_t v16bf;
typedef __attribute__((ext_vector_type(8)))  float  v8f;
typedef __attribute__((ext_vector_type(4)))  unsigned int v4u;
typedef __attribute__((ext_vector_type(8)))  int v8i;
typedef __attribute__((ext_vector_type(4)))  int v4i;

union BFrag { v16bf v; unsigned short u[16]; };
union AccF  { v8f   v; float f[8]; };

__device__ inline unsigned short f2bf(float f) {
  unsigned int u = __float_as_uint(f);
  unsigned int r = (u + 0x7FFFu + ((u >> 16) & 1u)) >> 16;
  return (unsigned short)r;
}
__device__ inline float gelu_exact(float x) {
  return 0.5f * x * (1.0f + erff(x * 0.70710678118654752f));
}

#if __has_builtin(__builtin_amdgcn_tensor_load_to_lds) && __has_builtin(__builtin_amdgcn_s_wait_tensorcnt)
#define USE_TDM 1
#else
#define USE_TDM 0
#endif

// Truncate a generic pointer to a __shared__ object to its 32-bit LDS offset
// (LDS aperture: addr[31:0] is the wave-relative LDS address, ISA 10.2).
__device__ inline unsigned lds_addr32(const void* p) {
  return (unsigned)(unsigned long long)p;
}

#if USE_TDM
// Issue one TDM 2D tile load: tile_rows x 32 bf16 elements from a row-major
// [rows, row_stride] bf16 tensor into LDS with a 1-DWORD pad every 16 DWORDs
// (=> 34-ushort LDS row stride). OOB rows/cols are hardware-zero-filled.
// This toolchain's builtin takes 6 args: (v4u, v8i, v4i, v4i, v8i, cpol).
__device__ inline void tdm_load_tile(const unsigned short* gbase, unsigned lds_off,
                                     unsigned rows_avail, unsigned cols_avail,
                                     unsigned row_stride_elems, unsigned tile_rows) {
  union { v4u v; unsigned int u[4]; } g0;
  union { v8i v; unsigned int u[8]; } g1, g4;
  union { v4i v; unsigned int u[4]; } g2, g3;
  unsigned long long ga = (unsigned long long)gbase;
  // D# group 0 (ISA 8.3): count=1, lds_addr, global_addr[56:0], type=2
  g0.u[0] = 1u;
  g0.u[1] = lds_off;
  g0.u[2] = (unsigned)(ga & 0xFFFFFFFFu);
  g0.u[3] = (unsigned)((ga >> 32) & 0x01FFFFFFu) | (2u << 30);
  // D# group 1 (ISA 8.4): data_size=1(2B), pad_enable, pad_interval=3(16 DW),
  // pad_amount=0(1 DW); tensor_dim0/1, tile_dim0=32, tile_dim1=tile_rows,
  // tensor_dim0_stride = row stride (elements).
  g1.u[0] = (1u << 16) | (1u << 20) | (3u << 22);
  g1.u[1] = (cols_avail & 0xFFFFu) << 16;
  g1.u[2] = (cols_avail >> 16) | ((rows_avail & 0xFFFFu) << 16);
  g1.u[3] = (rows_avail >> 16) | (32u << 16);
  g1.u[4] = tile_rows & 0xFFFFu;                // tile_dim1; tile_dim2=0
  g1.u[5] = row_stride_elems;                   // tensor_dim0_stride[31:0]
  g1.u[6] = 0u;                                 // stride hi / dim1_stride lo
  g1.u[7] = 0u;
  g2.u[0] = g2.u[1] = g2.u[2] = g2.u[3] = 0u;   // dims 2/3 unused
  g3.u[0] = g3.u[1] = g3.u[2] = g3.u[3] = 0u;
#pragma unroll
  for (int i = 0; i < 8; i++) g4.u[i] = 0u;
  __builtin_amdgcn_tensor_load_to_lds(g0.v, g1.v, g2.v, g3.v, g4.v, 0);
}
#endif

// =====================================================================
// Generic bf16 WMMA GEMM:  C[z] = act(A[z] (MxK) * W[z]^T (NxK) + bias[z])
// block tile 128x64, 8 waves of 32x32 WMMA tiles.
// K staged in 32-wide LDS slabs; TDM double-buffered DMA when available.
// =====================================================================
#define LDA_ 34
#define LDB_ 34
__global__ __launch_bounds__(256) void k_gemm_bf16(
    const unsigned short* __restrict__ A, const unsigned short* __restrict__ W,
    const float* __restrict__ bias,
    float* __restrict__ Cf, unsigned short* __restrict__ Cb,
    int Mdim, int Ndim, int Kdim, int act,
    long long batA, long long batW, long long batBias, long long batC)
{
  __shared__ unsigned short As[2][128 * LDA_];
  __shared__ unsigned short Bs[2][64 * LDB_];

  const int z = blockIdx.z;
  const unsigned short* Ab = A + (size_t)z * batA;
  const unsigned short* Wb = W + (size_t)z * batW;
  const float* biasz = bias + (size_t)z * batBias;

  const int n0 = blockIdx.x * 64;
  const int m0 = blockIdx.y * 128;
  const int t = threadIdx.x;
  const int wave = t >> 5, lane = t & 31;
  const int half = lane >> 4, r = lane & 15;
  const int waveM = (wave >> 1) * 32, waveN = (wave & 1) * 32;

  AccF acc[2][2];
#pragma unroll
  for (int i = 0; i < 2; i++)
#pragma unroll
    for (int j = 0; j < 2; j++)
#pragma unroll
      for (int v = 0; v < 8; v++) acc[i][j].f[v] = 0.0f;

#if USE_TDM
  // prologue: wave 0 posts DMA for K-slab 0 into buffer 0
  if (wave == 0) {
    tdm_load_tile(Ab + (size_t)m0 * Kdim, lds_addr32(&As[0][0]),
                  (unsigned)(Mdim - m0), (unsigned)Kdim, (unsigned)Kdim, 128u);
    tdm_load_tile(Wb + (size_t)n0 * Kdim, lds_addr32(&Bs[0][0]),
                  (unsigned)(Ndim - n0), (unsigned)Kdim, (unsigned)Kdim, 64u);
  }
#endif

  int buf = 0;
  for (int k0 = 0; k0 < Kdim; k0 += 32, buf ^= 1) {
#if USE_TDM
    const bool has_next = (k0 + 32 < Kdim);
    if (wave == 0 && has_next) {   // post DMA for next slab into other buffer
      tdm_load_tile(Ab + (size_t)m0 * Kdim + (k0 + 32), lds_addr32(&As[buf ^ 1][0]),
                    (unsigned)(Mdim - m0), (unsigned)(Kdim - (k0 + 32)),
                    (unsigned)Kdim, 128u);
      tdm_load_tile(Wb + (size_t)n0 * Kdim + (k0 + 32), lds_addr32(&Bs[buf ^ 1][0]),
                    (unsigned)(Ndim - n0), (unsigned)(Kdim - (k0 + 32)),
                    (unsigned)Kdim, 64u);
    }
    if (has_next) __builtin_amdgcn_s_wait_tensorcnt(2);  // current slab landed
    else          __builtin_amdgcn_s_wait_tensorcnt(0);
    __syncthreads();
    const unsigned short* At = &As[buf][0];
    const unsigned short* Bt = &Bs[buf][0];
#else
    __syncthreads();
    { // stage A tile 128x32 (each thread: 16 bf16 = 32B)
      int row = t >> 1, c0 = (t & 1) * 16;
      int m = m0 + row;
      union { uint4 q; unsigned short s[8]; } u0, u1;
      if (m < Mdim) {
        const uint4* p = (const uint4*)(Ab + (size_t)m * Kdim + k0 + c0);
        u0.q = p[0]; u1.q = p[1];
      } else {
#pragma unroll
        for (int i = 0; i < 8; i++) { u0.s[i] = 0; u1.s[i] = 0; }
      }
      unsigned short* d = &As[0][row * LDA_ + c0];
#pragma unroll
      for (int i = 0; i < 8; i++) { d[i] = u0.s[i]; d[8 + i] = u1.s[i]; }
      if (k0 + 32 < Kdim && m < Mdim)
        __builtin_prefetch(Ab + (size_t)m * Kdim + k0 + 32, 0, 1);
    }
    if (t < 128) { // stage W tile 64x32
      int row = t >> 1, c0 = (t & 1) * 16;
      int n = n0 + row;
      union { uint4 q; unsigned short s[8]; } u0, u1;
      if (n < Ndim) {
        const uint4* p = (const uint4*)(Wb + (size_t)n * Kdim + k0 + c0);
        u0.q = p[0]; u1.q = p[1];
        if (k0 + 32 < Kdim)
          __builtin_prefetch(Wb + (size_t)n * Kdim + k0 + 32, 0, 1);
      } else {
#pragma unroll
        for (int i = 0; i < 8; i++) { u0.s[i] = 0; u1.s[i] = 0; }
      }
      unsigned short* d = &Bs[0][row * LDB_ + c0];
#pragma unroll
      for (int i = 0; i < 8; i++) { d[i] = u0.s[i]; d[8 + i] = u1.s[i]; }
    }
    __syncthreads();
    const unsigned short* At = &As[0][0];
    const unsigned short* Bt = &Bs[0][0];
#endif

    // build fragments in contiguous k order (k = 16*half + s) for both A and B
    BFrag af[2], bfr[2];
#pragma unroll
    for (int i = 0; i < 2; i++) {
      const unsigned short* arow = &At[(waveM + i * 16 + r) * LDA_];
#pragma unroll
      for (int j = 0; j < 8; j++) {
        int k = half * 16 + 2 * j;
        af[i].u[2 * j] = arow[k]; af[i].u[2 * j + 1] = arow[k + 1];
      }
    }
#pragma unroll
    for (int i = 0; i < 2; i++) {
      const unsigned short* brow = &Bt[(waveN + i * 16 + r) * LDB_];
      int kb = half * 16;
#pragma unroll
      for (int j = 0; j < 8; j++) {
        bfr[i].u[2 * j] = brow[kb + 2 * j]; bfr[i].u[2 * j + 1] = brow[kb + 2 * j + 1];
      }
    }
#pragma unroll
    for (int i = 0; i < 2; i++)
#pragma unroll
      for (int j = 0; j < 2; j++)
        acc[i][j].v = __builtin_amdgcn_wmma_f32_16x16x32_bf16(
            false, af[i].v, false, bfr[j].v, (short)0, acc[i][j].v, false, false);

#if USE_TDM
    __syncthreads();   // all waves done with this buffer before next DMA reuses it
#endif
  }

  // epilogue: bias + activation + store (f32 and/or bf16)
#pragma unroll
  for (int i = 0; i < 2; i++) {
    int mbase = m0 + waveM + i * 16 + 8 * half;
#pragma unroll
    for (int j = 0; j < 2; j++) {
      int n = n0 + waveN + j * 16 + r;
      if (n >= Ndim) continue;
      float bv = biasz[n];
#pragma unroll
      for (int v = 0; v < 8; v++) {
        int m = mbase + v;
        if (m >= Mdim) continue;
        float x = acc[i][j].f[v] + bv;
        if (act == 1) x = gelu_exact(x);
        size_t idx = (size_t)z * batC + (size_t)m * Ndim + n;
        if (Cf) Cf[idx] = x;
        if (Cb) Cb[idx] = f2bf(x);
      }
    }
  }
}

// =====================================================================
// Fused attention per (b, h): scores = QK^T/8 + mask bias, softmax, P*V
// qkv bf16 [B*S, 3D]; output bf16 [B*S, D]. 8 waves, 16 query rows each.
// =====================================================================
__global__ __launch_bounds__(256) void k_attention(
    const unsigned short* __restrict__ qkv, const int* __restrict__ mask,
    unsigned short* __restrict__ attnout)
{
  __shared__ float sS[S_ * S_];   // 64 KB: scores then probabilities
  const int h = blockIdx.x, b = blockIdx.y;
  const int t = threadIdx.x, wave = t >> 5, lane = t & 31;
  const int half = lane >> 4, r = lane & 15;
  const size_t rs = 3 * D_;
  const unsigned short* base = qkv + (size_t)b * S_ * rs;
  const int qoff = h * DH_, koff = D_ + h * DH_, voff = 2 * D_ + h * DH_;

  // Q fragments for this wave's 16 query rows (dh=64 -> 2 K-steps of 32)
  BFrag qf[2];
#pragma unroll
  for (int ks = 0; ks < 2; ks++) {
    int m = wave * 16 + r;
    const unsigned short* qrow = base + (size_t)m * rs + qoff;
#pragma unroll
    for (int j = 0; j < 8; j++) {
      int k = ks * 32 + half * 16 + 2 * j;
      qf[ks].u[2 * j] = qrow[k]; qf[ks].u[2 * j + 1] = qrow[k + 1];
    }
  }
  // scores strip 16x128
  for (int nt = 0; nt < 8; nt++) {
    AccF acc;
#pragma unroll
    for (int v = 0; v < 8; v++) acc.f[v] = 0.0f;
#pragma unroll
    for (int ks = 0; ks < 2; ks++) {
      BFrag kf;
      int n = nt * 16 + r;
      const unsigned short* krow = base + (size_t)n * rs + koff;
      int kb = ks * 32 + half * 16;
#pragma unroll
      for (int j = 0; j < 8; j++) {
        kf.u[2 * j] = krow[kb + 2 * j]; kf.u[2 * j + 1] = krow[kb + 2 * j + 1];
      }
      acc.v = __builtin_amdgcn_wmma_f32_16x16x32_bf16(
          false, qf[ks].v, false, kf.v, (short)0, acc.v, false, false);
    }
    int col = nt * 16 + r;
    float biasv = (mask[b * S_ + col] == 0) ? -1e9f : 0.0f;
#pragma unroll
    for (int v = 0; v < 8; v++) {
      int row = wave * 16 + v + 8 * half;
      sS[row * S_ + col] = acc.f[v] * 0.125f + biasv;
    }
  }
  __syncthreads();
  if (t < S_) { // f32 softmax, one row per thread
    float* rp = &sS[t * S_];
    float mx = -1e30f;
    for (int i = 0; i < S_; i++) mx = fmaxf(mx, rp[i]);
    float sum = 0.0f;
    for (int i = 0; i < S_; i++) { float e = __expf(rp[i] - mx); rp[i] = e; sum += e; }
    float inv = 1.0f / sum;
    for (int i = 0; i < S_; i++) rp[i] *= inv;
  }
  __syncthreads();
  // O = P * V  (K=128 -> 4 K-steps; dh=64 -> 4 N-tiles)
  BFrag pf[4];
#pragma unroll
  for (int ks = 0; ks < 4; ks++) {
    int m = wave * 16 + r;
    const float* prow = &sS[m * S_];
#pragma unroll
    for (int j = 0; j < 8; j++) {
      int k = ks * 32 + half * 16 + 2 * j;
      pf[ks].u[2 * j] = f2bf(prow[k]); pf[ks].u[2 * j + 1] = f2bf(prow[k + 1]);
    }
  }
  for (int nt = 0; nt < 4; nt++) {
    AccF acc;
#pragma unroll
    for (int v = 0; v < 8; v++) acc.f[v] = 0.0f;
#pragma unroll
    for (int ks = 0; ks < 4; ks++) {
      BFrag vf;
      int n = nt * 16 + r;
      int kb = ks * 32 + half * 16;
#pragma unroll
      for (int j = 0; j < 8; j++) {
        vf.u[2 * j]     = base[(size_t)(kb + 2 * j) * rs + voff + n];
        vf.u[2 * j + 1] = base[(size_t)(kb + 2 * j + 1) * rs + voff + n];
      }
      acc.v = __builtin_amdgcn_wmma_f32_16x16x32_bf16(
          false, pf[ks].v, false, vf.v, (short)0, acc.v, false, false);
    }
#pragma unroll
    for (int v = 0; v < 8; v++) {
      int row = wave * 16 + v + 8 * half;
      attnout[((size_t)b * S_ + row) * D_ + h * DH_ + nt * 16 + r] = f2bf(acc.f[v]);
    }
  }
}

// =====================================================================
// Fused (residual +) LayerNorm (+ post-add + relu); f32 + bf16 outputs.
// =====================================================================
__global__ __launch_bounds__(256) void k_ln(
    const float* __restrict__ in, const float* __restrict__ resid,
    const float* __restrict__ g0, const float* __restrict__ b0, int gamma_per,
    const float* __restrict__ post, int post_per, int post_relu,
    float* __restrict__ outf, unsigned short* __restrict__ outb, int L)
{
  __shared__ float red[256];
  const int row = blockIdx.x, t = threadIdx.x;
  const int e = L >> 8;                       // 4 or 8
  const size_t goff = gamma_per ? (size_t)(row / gamma_per) * L : 0;
  const float* g = g0 + goff;
  const float* bb = b0 + goff;
  const size_t rb = (size_t)row * L;

  float vals[8];
  float s = 0.0f;
  for (int i = 0; i < e; i++) {
    int c = t + i * 256;
    float v = in[rb + c];
    if (resid) v += resid[rb + c];
    vals[i] = v; s += v;
  }
  red[t] = s; __syncthreads();
  for (int o = 128; o > 0; o >>= 1) { if (t < o) red[t] += red[t + o]; __syncthreads(); }
  float mean = red[0] / (float)L;
  __syncthreads();
  float vs = 0.0f;
  for (int i = 0; i < e; i++) { float d = vals[i] - mean; vs += d * d; }
  red[t] = vs; __syncthreads();
  for (int o = 128; o > 0; o >>= 1) { if (t < o) red[t] += red[t + o]; __syncthreads(); }
  float rstd = rsqrtf(red[0] / (float)L + 1e-5f);
  const float* pp = post ? post + (size_t)(row % post_per) * L : nullptr;
  for (int i = 0; i < e; i++) {
    int c = t + i * 256;
    float y = (vals[i] - mean) * rstd * g[c] + bb[c];
    if (pp) y += pp[c];
    if (post_relu) y = fmaxf(y, 0.0f);
    if (outf) outf[rb + c] = y;
    if (outb) outb[rb + c] = f2bf(y);
  }
}

// ---------------- small kernels ----------------
__global__ __launch_bounds__(256) void k_embed(
    const int* __restrict__ ids, const float* __restrict__ tok,
    const float* __restrict__ pos, float* __restrict__ xf,
    unsigned short* __restrict__ xb)
{
  int rsx = blockIdx.x;                 // b*S + s
  int sIdx = rsx & (S_ - 1);
  int id = ids[rsx];
  int t = threadIdx.x;
#pragma unroll
  for (int i = 0; i < 4; i++) {
    int c = t + i * 256;
    float v = tok[(size_t)id * D_ + c] + pos[(size_t)sIdx * D_ + c];
    xf[(size_t)rsx * D_ + c] = v;
    xb[(size_t)rsx * D_ + c] = f2bf(v);
  }
}

__global__ __launch_bounds__(256) void k_pool(
    const float* __restrict__ x, const int* __restrict__ mask,
    float* __restrict__ pooled, unsigned short* __restrict__ pooledb)
{
  int b = blockIdx.x, t = threadIdx.x;
  float acc[4] = {0.f, 0.f, 0.f, 0.f};
  float msum = 0.0f;
  for (int s = 0; s < S_; s++) {
    float m = (float)mask[b * S_ + s];
    msum += m;
    const float* rp = x + ((size_t)b * S_ + s) * D_;
#pragma unroll
    for (int i = 0; i < 4; i++) acc[i] += rp[t + i * 256] * m;
  }
  float inv = 1.0f / fmaxf(msum, 1e-9f);
#pragma unroll
  for (int i = 0; i < 4; i++) {
    float v = acc[i] * inv;
    pooled [(size_t)b * D_ + t + i * 256] = v;
    pooledb[(size_t)b * D_ + t + i * 256] = f2bf(v);
  }
}

__global__ __launch_bounds__(32) void k_gate(
    const float* __restrict__ pooled, const float* __restrict__ gw,
    const float* __restrict__ gb, float* __restrict__ gate)
{
  int b = blockIdx.x, lane = threadIdx.x;
  float dot = -1e30f;
  if (lane < E_) {
    const float* p = pooled + (size_t)b * D_;
    const float* w = gw + (size_t)lane * D_;
    float d = 0.0f;
    for (int i = 0; i < D_; i++) d += p[i] * w[i];
    dot = d + gb[lane];
  }
  float mx = dot;
  for (int m = 8; m > 0; m >>= 1) mx = fmaxf(mx, __shfl_xor(mx, m, 32));
  float ex = __expf(dot - mx);
  float sum = ex;
  for (int m = 8; m > 0; m >>= 1) sum += __shfl_xor(sum, m, 32);
  if (lane < E_) gate[(size_t)b * E_ + lane] = ex / sum;
}

__global__ __launch_bounds__(256) void k_combine(
    const float* __restrict__ gate, const float* __restrict__ eo,
    float* __restrict__ out)
{
  int b = blockIdx.x, t = threadIdx.x;
  for (int o = t; o < O_; o += 256) {
    float acc = 0.0f;
#pragma unroll
    for (int e = 0; e < E_; e++)
      acc += gate[(size_t)b * E_ + e] * eo[((size_t)e * B_ + b) * O_ + o];
    out[(size_t)b * O_ + o] = acc;
  }
}

__global__ __launch_bounds__(256) void k_f2bf(
    const float* __restrict__ src, unsigned short* __restrict__ dst, long long n)
{
  long long i = (long long)blockIdx.x * blockDim.x + threadIdx.x;
  long long stride = (long long)gridDim.x * blockDim.x;
  for (; i < n; i += stride) dst[i] = f2bf(src[i]);
}

// =====================================================================
// Host orchestration
// =====================================================================
extern "C" void kernel_launch(void* const* d_in, const int* in_sizes, int n_in,
                              void* d_out, int out_size, void* d_ws, size_t ws_size,
                              hipStream_t stream)
{
  (void)in_sizes; (void)n_in; (void)out_size; (void)ws_size;

  const int*   ids  = (const int*)d_in[0];
  const int*   mask = (const int*)d_in[1];
  const float* tok  = (const float*)d_in[2];
  const float* pos  = (const float*)d_in[3];
  struct Layer { const float *wqkv,*bqkv,*wo,*bo,*ln1g,*ln1b,*w1,*b1,*w2,*b2,*ln2g,*ln2b; } Lp[2];
  int ix = 4;
  for (int l = 0; l < 2; l++) {
    Lp[l].wqkv=(const float*)d_in[ix++]; Lp[l].bqkv=(const float*)d_in[ix++];
    Lp[l].wo  =(const float*)d_in[ix++]; Lp[l].bo  =(const float*)d_in[ix++];
    Lp[l].ln1g=(const float*)d_in[ix++]; Lp[l].ln1b=(const float*)d_in[ix++];
    Lp[l].w1  =(const float*)d_in[ix++]; Lp[l].b1  =(const float*)d_in[ix++];
    Lp[l].w2  =(const float*)d_in[ix++]; Lp[l].b2  =(const float*)d_in[ix++];
    Lp[l].ln2g=(const float*)d_in[ix++]; Lp[l].ln2b=(const float*)d_in[ix++];
  }
  const float* lnfg =(const float*)d_in[ix++]; const float* lnfb =(const float*)d_in[ix++];
  const float* gw   =(const float*)d_in[ix++]; const float* gb   =(const float*)d_in[ix++];
  const float* efc1w=(const float*)d_in[ix++]; const float* efc1b=(const float*)d_in[ix++];
  const float* eln1g=(const float*)d_in[ix++]; const float* eln1b=(const float*)d_in[ix++];
  const float* efc2w=(const float*)d_in[ix++]; const float* efc2b=(const float*)d_in[ix++];
  const float* eln2g=(const float*)d_in[ix++]; const float* eln2b=(const float*)d_in[ix++];
  const float* eoutw=(const float*)d_in[ix++]; const float* eoutb=(const float*)d_in[ix++];

  // ---- workspace bump allocation (phase-aliased) ----
  char* ws = (char*)d_ws;
  size_t off = 0;
  auto alloc = [&](size_t bytes) -> char* {
    char* p = ws + off; off += (bytes + 255) & ~(size_t)255; return p;
  };
  const size_t Nx = (size_t)MTOK * D_;
  float*          xf   = (float*)         alloc(Nx * 4);
  unsigned short* xb   = (unsigned short*)alloc(Nx * 2);
  unsigned short* qkvb = (unsigned short*)alloc((size_t)MTOK * 3 * D_ * 2);
  unsigned short* attb = (unsigned short*)alloc(Nx * 2);
  float*          of   = (float*)         alloc(Nx * 4);
  unsigned short* hb   = (unsigned short*)alloc((size_t)MTOK * FF_ * 2);
  unsigned short* wq16[2], *wo16[2], *w116[2], *w216[2];
  for (int l = 0; l < 2; l++) {
    wq16[l] = (unsigned short*)alloc((size_t)3 * D_ * D_ * 2);
    wo16[l] = (unsigned short*)alloc((size_t)D_ * D_ * 2);
    w116[l] = (unsigned short*)alloc((size_t)FF_ * D_ * 2);
    w216[l] = (unsigned short*)alloc((size_t)D_ * FF_ * 2);
  }
  // expert-phase aliases (regions dead by then)
  unsigned short* efc1w16 = qkvb;                                  // 67 MB
  unsigned short* efc2w16 = efc1w16 + (size_t)E_ * 2 * D_ * D_;    // 67 MB
  unsigned short* eoutw16 = efc2w16 + (size_t)E_ * D_ * 2 * D_;    // 33 MB (fits in 201 MB)
  float*          efc1f   = (float*)attb;                          // E*B*2D f32 (33.5 MB)
  unsigned short* h1b     = (unsigned short*)of;                   // E*B*2D bf16
  float*          efc2f   = (float*)((char*)of + (size_t)E_ * B_ * 2 * D_ * 2);
  unsigned short* h2b     = (unsigned short*)((char*)efc2f + (size_t)E_ * B_ * D_ * 4);
  float*          eof     = (float*)((char*)h2b + (size_t)E_ * B_ * D_ * 2);
  float*          pooled  = (float*)hb;
  unsigned short* pooledb = (unsigned short*)((char*)hb + (size_t)B_ * D_ * 4);
  float*          gatep   = (float*)((char*)hb + (size_t)B_ * D_ * 6);

  auto gemm = [&](const unsigned short* A, const unsigned short* W, const float* bias,
                  float* Cf, unsigned short* Cb, int M, int N, int K, int act,
                  int batch, long long bA, long long bW, long long bB, long long bC) {
    dim3 g((unsigned)((N + 63) / 64), (unsigned)((M + 127) / 128), (unsigned)batch);
    k_gemm_bf16<<<g, 256, 0, stream>>>(A, W, bias, Cf, Cb, M, N, K, act, bA, bW, bB, bC);
  };

  // ---- embedding + layer weight conversion ----
  k_embed<<<dim3(MTOK), 256, 0, stream>>>(ids, tok, pos, xf, xb);
  for (int l = 0; l < 2; l++) {
    k_f2bf<<<2048, 256, 0, stream>>>(Lp[l].wqkv, wq16[l], (long long)3 * D_ * D_);
    k_f2bf<<<2048, 256, 0, stream>>>(Lp[l].wo,   wo16[l], (long long)D_ * D_);
    k_f2bf<<<2048, 256, 0, stream>>>(Lp[l].w1,   w116[l], (long long)FF_ * D_);
    k_f2bf<<<2048, 256, 0, stream>>>(Lp[l].w2,   w216[l], (long long)D_ * FF_);
  }

  // ---- transformer layers (post-LN) ----
  for (int l = 0; l < 2; l++) {
    gemm(xb, wq16[l], Lp[l].bqkv, nullptr, qkvb, MTOK, 3 * D_, D_, 0, 1, 0, 0, 0, 0);
    k_attention<<<dim3(NH_, B_), 256, 0, stream>>>(qkvb, mask, attb);
    gemm(attb, wo16[l], Lp[l].bo, of, nullptr, MTOK, D_, D_, 0, 1, 0, 0, 0, 0);
    k_ln<<<dim3(MTOK), 256, 0, stream>>>(xf, of, Lp[l].ln1g, Lp[l].ln1b, 0,
                                         nullptr, 1, 0, xf, xb, D_);
    gemm(xb, w116[l], Lp[l].b1, nullptr, hb, MTOK, FF_, D_, 1, 1, 0, 0, 0, 0);
    gemm(hb, w216[l], Lp[l].b2, of, nullptr, MTOK, D_, FF_, 0, 1, 0, 0, 0, 0);
    k_ln<<<dim3(MTOK), 256, 0, stream>>>(xf, of, Lp[l].ln2g, Lp[l].ln2b, 0,
                                         nullptr, 1, 0, xf, xb, D_);
  }

  // ---- final LN, masked mean pool, gate ----
  k_ln<<<dim3(MTOK), 256, 0, stream>>>(xf, nullptr, lnfg, lnfb, 0,
                                       nullptr, 1, 0, of, nullptr, D_);
  k_pool<<<dim3(B_), 256, 0, stream>>>(of, mask, pooled, pooledb);
  k_gate<<<dim3(B_), 32, 0, stream>>>(pooled, gw, gb, gatep);

  // ---- expert stack (batched over E) ----
  k_f2bf<<<4096, 256, 0, stream>>>(efc1w, efc1w16, (long long)E_ * 2 * D_ * D_);
  k_f2bf<<<4096, 256, 0, stream>>>(efc2w, efc2w16, (long long)E_ * D_ * 2 * D_);
  k_f2bf<<<4096, 256, 0, stream>>>(eoutw, eoutw16, (long long)E_ * O_ * D_);

  gemm(pooledb, efc1w16, efc1b, efc1f, nullptr, B_, 2 * D_, D_, 0,
       E_, 0, (long long)2 * D_ * D_, 2 * D_, (long long)B_ * 2 * D_);
  k_ln<<<dim3(E_ * B_), 256, 0, stream>>>(efc1f, nullptr, eln1g, eln1b, B_,
                                          nullptr, 1, 1, nullptr, h1b, 2 * D_);
  gemm(h1b, efc2w16, efc2b, efc2f, nullptr, B_, D_, 2 * D_, 0,
       E_, (long long)B_ * 2 * D_, (long long)D_ * 2 * D_, D_, (long long)B_ * D_);
  k_ln<<<dim3(E_ * B_), 256, 0, stream>>>(efc2f, nullptr, eln2g, eln2b, B_,
                                          pooled, B_, 1, nullptr, h2b, D_);
  gemm(h2b, eoutw16, eoutb, eof, nullptr, B_, O_, D_, 0,
       E_, (long long)B_ * D_, (long long)O_ * D_, O_, (long long)B_ * O_);

  k_combine<<<dim3(B_), 256, 0, stream>>>(gatep, eof, (float*)d_out);
}
